// Head_30605936951765
// MI455X (gfx1250) — matrix operand order repaired and checked by reading, not compiled, not run on hardware
//
#include <hip/hip_runtime.h>

// ---- MI455X / gfx1250, wave32, WMMA bf16 path ----
#define N_E 1024
#define SEQ 4096
#define NB  4
#define DH  64
#define ROWS (NB * SEQ)          // 16384 flat rows
#define TILES (ROWS / 16)        // 1024 16-row tiles
#define WAVES_PER_BLOCK 8
#define BLOCKS (TILES / WAVES_PER_BLOCK)  // 128
#define WELEM (DH * N_E)         // 65536 elements per weight matrix

typedef __attribute__((ext_vector_type(2)))  float          v2f;
typedef __attribute__((ext_vector_type(4)))  float          v4f;
typedef __attribute__((ext_vector_type(8)))  float          v8f;
typedef __attribute__((ext_vector_type(16))) float          v16f;
typedef __attribute__((ext_vector_type(2)))  __bf16         v2bf;
typedef __attribute__((ext_vector_type(4)))  __bf16         v4bf;
typedef __attribute__((ext_vector_type(8)))  __bf16         v8bf;
typedef __attribute__((ext_vector_type(16))) __bf16         v16bf;
typedef __attribute__((ext_vector_type(16))) unsigned short v16u;
typedef __attribute__((ext_vector_type(8)))  unsigned short v8u;
typedef __attribute__((ext_vector_type(2)))  unsigned int   v2ui;
typedef __attribute__((ext_vector_type(4)))  unsigned int   v4ui;

// f32 -> bf16 conversions through native fptrunc (backend may select
// v_cvt_pk_bf16_f32; falls back to RNE expansion otherwise).
__device__ __forceinline__ unsigned int cvt2(float a, float b) {
    v2f t; t[0] = a; t[1] = b;
    v2bf p = __builtin_convertvector(t, v2bf);
    return __builtin_bit_cast(unsigned int, p);
}
__device__ __forceinline__ unsigned short cvt1(float a) {
    return __builtin_bit_cast(unsigned short, (__bf16)a);
}

__device__ __forceinline__ v16bf cat8(v8u lo, v8u hi) {
    v16u u;
#pragma unroll
    for (int w = 0; w < 8; ++w) { u[w] = lo[w]; u[8 + w] = hi[w]; }
    return __builtin_bit_cast(v16bf, u);
}

// reductions across the 16 lanes of a column group (xor masks < 16 stay in-group)
__device__ __forceinline__ float redmax16(float v) {
#pragma unroll
    for (int xm = 1; xm < 16; xm <<= 1) v = fmaxf(v, __shfl_xor(v, xm, 32));
    return v;
}
__device__ __forceinline__ float redsum16(float v) {
#pragma unroll
    for (int xm = 1; xm < 16; xm <<= 1) v += __shfl_xor(v, xm, 32);
    return v;
}

// ---------------------------------------------------------------------------
// Kernel 0: one-shot weight conversion f32 -> bf16 (Wb = [Wq | Wk | Wv]).
// ---------------------------------------------------------------------------
__global__ __launch_bounds__(256) void wcvt_kernel(
    const float* __restrict__ Wq, const float* __restrict__ Wk,
    const float* __restrict__ Wv, unsigned short* __restrict__ Wb)
{
    const int mt  = blockIdx.x >> 6;              // 64 blocks per matrix
    const int blk = blockIdx.x & 63;
    const float* src = (mt == 0) ? Wq : (mt == 1) ? Wk : Wv;
    const int idx = (blk * 256 + threadIdx.x) * 4;  // 64*256*4 = 65536
    v4f f = *(const v4f*)(src + idx);
    v4bf c = __builtin_convertvector(f, v4bf);
    *(v2ui*)(Wb + (size_t)mt * WELEM + idx) = __builtin_bit_cast(v2ui, c);
}

// ---------------------------------------------------------------------------
// Kernel 1: QKV projection.  One wave -> one 16-row tile of X.
// Q,K stored row-major bf16 [16384][64]; V stored transposed bf16 [b][64][4096].
// ---------------------------------------------------------------------------
__global__ __launch_bounds__(256) void qkv_proj_kernel(
    const float* __restrict__ X,
    const unsigned short* __restrict__ Wb,
    const float* __restrict__ bq, const float* __restrict__ bk,
    const float* __restrict__ bv,
    unsigned short* __restrict__ Qb, unsigned short* __restrict__ Kb,
    unsigned short* __restrict__ VTb)
{
    const int lane = threadIdx.x & 31;
    const int wave = threadIdx.x >> 5;
    const int tile = blockIdx.x * WAVES_PER_BLOCK + wave;   // 0..1023
    const int row0 = tile * 16;
    const int m = lane & 15;
    const int h = lane >> 4;

    const float* bias[3] = { bq, bk, bv };

    v8f acc[3][4] = {};

#pragma unroll 2
    for (int k0 = 0; k0 < N_E; k0 += 32) {
        // A fragment: X rows f32 -> bf16, element e -> K = 16*(e/8) + 8*h + e%8
        v16f av;
        const float* xrow = X + (size_t)(row0 + m) * N_E + k0 + 8 * h;
#pragma unroll
        for (int g = 0; g < 2; ++g) {
            v4f f0 = *(const v4f*)(xrow + g * 16);
            v4f f1 = *(const v4f*)(xrow + g * 16 + 4);
#pragma unroll
            for (int w = 0; w < 4; ++w) { av[g * 8 + w] = f0[w]; av[g * 8 + 4 + w] = f1[w]; }
        }
        v16bf a = __builtin_convertvector(av, v16bf);

#pragma unroll
        for (int mt = 0; mt < 3; ++mt) {
#pragma unroll
            for (int nt = 0; nt < 4; ++nt) {
                // B fragment straight from pre-converted bf16 weights:
                // B[dd][n] = W[n][dd]; lane col n=m, element e -> dd = k0+16*(e/8)+8h+e%8
                const unsigned short* wrow =
                    Wb + (size_t)mt * WELEM + (size_t)(nt * 16 + m) * N_E + k0 + 8 * h;
                v8u lo = *(const v8u*)wrow;
                v8u hi = *(const v8u*)(wrow + 16);
                v16bf bf = cat8(lo, hi);
                acc[mt][nt] = __builtin_amdgcn_wmma_f32_16x16x32_bf16(
                    false, a, false, bf, (short)0, acc[mt][nt], false, false);
            }
        }
    }

    const int b  = row0 >> 12;     // / SEQ
    const int s0 = row0 & (SEQ - 1);
#pragma unroll
    for (int mt = 0; mt < 3; ++mt) {
#pragma unroll
        for (int nt = 0; nt < 4; ++nt) {
            const int col = nt * 16 + m;                 // C layout: col = lane%16
            const float bb = bias[mt][col];
            v8f c = acc[mt][nt];
            if (mt < 2) {
                unsigned short* dst = (mt == 0) ? Qb : Kb;
#pragma unroll
                for (int r = 0; r < 8; ++r) {            // row = r + 8*h
                    dst[(size_t)(row0 + r + 8 * h) * DH + col] = cvt1(c[r] + bb);
                }
            } else {
                // V^T[b][col][s]: 8 consecutive s per lane -> one b128 store
                v8f vv;
#pragma unroll
                for (int r = 0; r < 8; ++r) vv[r] = c[r] + bb;
                v8bf vb = __builtin_convertvector(vv, v8bf);
                *(v4ui*)(VTb + ((size_t)(b * DH + col) * SEQ + s0 + 8 * h)) =
                    __builtin_bit_cast(v4ui, vb);
            }
        }
    }
}

// ---------------------------------------------------------------------------
// Kernel 2: causal flash attention. One wave -> one 16-query tile, iterating
// 32-key blocks up to the causal frontier. Online softmax, WMMA bf16.
// ---------------------------------------------------------------------------
__global__ __launch_bounds__(256) void attn_kernel(
    const unsigned short* __restrict__ Qb,
    const unsigned short* __restrict__ Kb,
    const unsigned short* __restrict__ VTb,
    float* __restrict__ Out)
{
    __shared__ __align__(16) unsigned short pbuf[WAVES_PER_BLOCK][16 * 32];

    const int lane = threadIdx.x & 31;
    const int wave = threadIdx.x >> 5;
    const int tile = blockIdx.x * WAVES_PER_BLOCK + wave;   // 0..1023
    const int row0 = tile * 16;
    const int b   = row0 >> 12;
    const int sq0 = row0 & (SEQ - 1);
    const int m = lane & 15;
    const int h = lane >> 4;
    unsigned short* pb = pbuf[wave];

    // Q A-fragments, resident for whole kernel (2 d-steps of 32)
    v16bf qf[2];
    {
        const unsigned short* qrow = Qb + (size_t)(row0 + m) * DH + 8 * h;
#pragma unroll
        for (int ks = 0; ks < 2; ++ks) {
            v8u lo = *(const v8u*)(qrow + ks * 32);
            v8u hi = *(const v8u*)(qrow + ks * 32 + 16);
            qf[ks] = cat8(lo, hi);
        }
    }

    float mrun[8], lrun[8];
#pragma unroll
    for (int r = 0; r < 8; ++r) { mrun[r] = -__builtin_inff(); lrun[r] = 0.f; }
    v8f o[4] = {};

    const size_t kbase = (size_t)b * SEQ;

    for (int j0 = 0; j0 <= sq0; j0 += 32) {
        // --- preload all 4 K fragments (8 clause-able b128 loads) ---
        v16bf kf[2][2];
#pragma unroll
        for (int t = 0; t < 2; ++t)
#pragma unroll
            for (int ks = 0; ks < 2; ++ks) {
                const unsigned short* krow =
                    Kb + (kbase + j0 + 16 * t + m) * DH + ks * 32 + 8 * h;
                v8u lo = *(const v8u*)krow;
                v8u hi = *(const v8u*)(krow + 16);
                kf[t][ks] = cat8(lo, hi);
            }
        // --- S = Q * K^T for 32 keys: two 16x16 C tiles, K=64 over d ---
        v8f st[2] = {};
#pragma unroll
        for (int t = 0; t < 2; ++t)
#pragma unroll
            for (int ks = 0; ks < 2; ++ks)
                st[t] = __builtin_amdgcn_wmma_f32_16x16x32_bf16(
                    false, qf[ks], false, kf[t][ks], (short)0, st[t], false, false);

        // --- scale + causal mask (col key = j0+16t+m ; row q = sq0+r+8h) ---
#pragma unroll
        for (int t = 0; t < 2; ++t) {
            const int key = j0 + 16 * t + m;
#pragma unroll
            for (int r = 0; r < 8; ++r) {
                const int q = sq0 + r + 8 * h;
                float v = st[t][r] * 0.125f;
                st[t][r] = (key <= q) ? v : -__builtin_inff();
            }
        }
        // --- online softmax stats (row reductions across 16-lane groups) ---
        float alpha[8];
#pragma unroll
        for (int r = 0; r < 8; ++r) {
            float rmax = redmax16(fmaxf(st[0][r], st[1][r]));
            float mn = fmaxf(mrun[r], rmax);
            alpha[r] = __expf(mrun[r] - mn);
            mrun[r] = mn;
        }
#pragma unroll
        for (int t = 0; t < 2; ++t)
#pragma unroll
            for (int r = 0; r < 8; ++r)
                st[t][r] = __expf(st[t][r] - mrun[r]);
#pragma unroll
        for (int r = 0; r < 8; ++r) {
            float rsum = redsum16(st[0][r] + st[1][r]);
            lrun[r] = lrun[r] * alpha[r] + rsum;
        }
#pragma unroll
        for (int nt = 0; nt < 4; ++nt)
#pragma unroll
            for (int r = 0; r < 8; ++r)
                o[nt][r] *= alpha[r];

        // --- P: C-layout -> A-layout via per-wave LDS staging ---
#pragma unroll
        for (int r = 0; r < 8; ++r) {
            unsigned int pk = cvt2(st[0][r], st[1][r]);  // (t=0, t=1) pair
            pb[(r + 8 * h) * 32 + m]      = (unsigned short)(pk & 0xFFFFu);
            pb[(r + 8 * h) * 32 + 16 + m] = (unsigned short)(pk >> 16);
        }
        __builtin_amdgcn_wave_barrier();       // DS ops in-order per wave
        v16bf pf;
        {
            const unsigned short* prow = pb + m * 32 + 8 * h;
            v8u lo = *(const v8u*)prow;
            v8u hi = *(const v8u*)(prow + 16);
            pf = cat8(lo, hi);
        }
        __builtin_amdgcn_wave_barrier();

        // --- preload all 4 V fragments, then O += P * V ---
        v16bf vf[4];
#pragma unroll
        for (int nt = 0; nt < 4; ++nt) {
            const unsigned short* vrow =
                VTb + (size_t)(b * DH + nt * 16 + m) * SEQ + j0 + 8 * h;
            v8u lo = *(const v8u*)vrow;
            v8u hi = *(const v8u*)(vrow + 16);
            vf[nt] = cat8(lo, hi);
        }
#pragma unroll
        for (int nt = 0; nt < 4; ++nt)
            o[nt] = __builtin_amdgcn_wmma_f32_16x16x32_bf16(
                false, pf, false, vf[nt], (short)0, o[nt], false, false);

        if (j0 + 32 <= sq0)
            __builtin_prefetch(Kb + (kbase + j0 + 32 + m) * DH, 0, 1);
    }

    // --- finalize: O / l, fp32 out [b][s][64] ---
#pragma unroll
    for (int nt = 0; nt < 4; ++nt)
#pragma unroll
        for (int r = 0; r < 8; ++r)
            Out[(size_t)(row0 + r + 8 * h) * DH + nt * 16 + m] = o[nt][r] / lrun[r];
}

extern "C" void kernel_launch(void* const* d_in, const int* in_sizes, int n_in,
                              void* d_out, int out_size, void* d_ws, size_t ws_size,
                              hipStream_t stream) {
    (void)in_sizes; (void)n_in; (void)out_size; (void)ws_size;
    const float* X  = (const float*)d_in[0];
    const float* Wq = (const float*)d_in[1];
    const float* bq = (const float*)d_in[2];
    const float* Wk = (const float*)d_in[3];
    const float* bk = (const float*)d_in[4];
    const float* Wv = (const float*)d_in[5];
    const float* bv = (const float*)d_in[6];

    unsigned short* Qb  = (unsigned short*)d_ws;               // 2 MB
    unsigned short* Kb  = Qb + (size_t)ROWS * DH;              // 2 MB
    unsigned short* VTb = Kb + (size_t)ROWS * DH;              // 2 MB
    unsigned short* Wbb = VTb + (size_t)ROWS * DH;             // 384 KB

    wcvt_kernel<<<192, 256, 0, stream>>>(Wq, Wk, Wv, Wbb);
    qkv_proj_kernel<<<BLOCKS, 256, 0, stream>>>(X, Wbb, bq, bk, bv, Qb, Kb, VTb);
    attn_kernel<<<BLOCKS, 256, 0, stream>>>(Qb, Kb, VTb, (float*)d_out);
}
